// HybridModel_35502199669531
// MI455X (gfx1250) — matrix operand order repaired and checked by reading, not compiled, not run on hardware
//
#include <hip/hip_runtime.h>
#include <hip/hip_bf16.h>
#include <math.h>

// ---------- shapes ----------
#define NROW 8192
#define DIMD 512
#define HDIM 64
#define WPW  4          // waves per workgroup in sim kernel

// ---------- vector types ----------
typedef __bf16 bf16x8  __attribute__((ext_vector_type(8)));
typedef __bf16 bf16x16 __attribute__((ext_vector_type(16)));
typedef float  f32x8   __attribute__((ext_vector_type(8)));
typedef float  f32x16  __attribute__((ext_vector_type(16)));

// ---------- small helpers ----------
__device__ __forceinline__ void insert5(float* v5, int* i5, float v, int ix) {
    if (v <= v5[4]) return;
    v5[4] = v; i5[4] = ix;
#pragma unroll
    for (int a = 4; a > 0; --a) {
        if (v5[a] > v5[a-1]) {
            float tf = v5[a]; v5[a] = v5[a-1]; v5[a-1] = tf;
            int   tq = i5[a]; i5[a] = i5[a-1]; i5[a-1] = tq;
        }
    }
}

// ============================================================
// 1) Row-normalize X -> bf16 Xn  (wave per row)
// ============================================================
__global__ __launch_bounds__(128) void rownorm_kernel(const float* __restrict__ X,
                                                      __bf16* __restrict__ Xn) {
    const int row  = blockIdx.x * 4 + (threadIdx.x >> 5);
    const int lane = threadIdx.x & 31;
    const float* xr = X + (size_t)row * DIMD;
    float ss = 0.f;
    for (int k = lane; k < DIMD; k += 32) { float v = xr[k]; ss += v * v; }
#pragma unroll
    for (int off = 16; off; off >>= 1) ss += __shfl_xor(ss, off, 32);
    const float inv = rsqrtf(ss);
    __bf16* orow = Xn + (size_t)row * DIMD;
    for (int k = lane; k < DIMD; k += 32) orow[k] = (__bf16)(xr[k] * inv);
}

// ============================================================
// 2) Fused sim = Xn @ Xn.T  + streaming exact top-5 per row
//    grid: NROW/16 blocks, 128 threads (4 waves), bf16 WMMA
//    Double-buffered B fragments: next tile's 32 b128 loads are
//    in flight while the current tile's 16 WMMAs + top-k run.
// ============================================================
__global__ __launch_bounds__(128) void sim_topk_kernel(const __bf16* __restrict__ Xn,
                                                       int* __restrict__ topk_idx,
                                                       float* __restrict__ topk_w) {
    __shared__ float sv[16][WPW * 16 * 5];
    __shared__ int   si[16][WPW * 16 * 5];

    const int lane    = threadIdx.x & 31;
    const int wave    = threadIdx.x >> 5;
    const int i0      = blockIdx.x * 16;
    const int halfsel = lane >> 4;       // 0 or 1
    const int l15     = lane & 15;

    // Preload A fragments for this 16-row strip (K = 512 -> 16 chunks of 32)
    const int khalf = halfsel ? 16 : 0;
    const __bf16* arow = Xn + (size_t)(i0 + l15) * DIMD + khalf;
    bf16x16 afr[16];
#pragma unroll
    for (int kc = 0; kc < 16; ++kc) {
        bf16x8 lo = *(const bf16x8*)(arow + kc * 32);
        bf16x8 hi = *(const bf16x8*)(arow + kc * 32 + 8);
        afr[kc] = __builtin_shufflevector(lo, hi, 0,1,2,3,4,5,6,7,8,9,10,11,12,13,14,15);
    }

    float tv[8][5]; int ti[8][5];
#pragma unroll
    for (int r = 0; r < 8; ++r)
#pragma unroll
        for (int e = 0; e < 5; ++e) { tv[r][e] = -1e30f; ti[r][e] = 0; }

    const int bkoff = halfsel ? 16 : 0;
    const int rbase = i0 + (halfsel ? 8 : 0);

    auto loadB = [&](int jtile, bf16x16* dst) {
        const __bf16* brow = Xn + (size_t)(jtile * 16 + l15) * DIMD + bkoff;
#pragma unroll
        for (int kc = 0; kc < 16; ++kc) dst[kc] = *(const bf16x16*)(brow + kc * 32);
    };
    auto compute = [&](const bf16x16* b, int j0) {
        f32x8 c = {0.f,0.f,0.f,0.f,0.f,0.f,0.f,0.f};
#pragma unroll
        for (int kc = 0; kc < 16; ++kc)
            c = __builtin_amdgcn_wmma_f32_16x16x32_bf16(false, afr[kc], false, b[kc],
                                                        (short)0, c, false, false);
        const int col = j0 + l15;
#pragma unroll
        for (int r = 0; r < 8; ++r) {
            if (rbase + r != col) insert5(tv[r], ti[r], c[r], col);   // exclude self
        }
    };

    const int NTJ = NROW / 16;           // 512 tiles
    bf16x16 bA[16], bB[16];
    loadB(wave, bA);
    // two tiles per iteration; jt+WPW always < NTJ (wave<4, NTJ%8==0)
    for (int jt = wave; jt < NTJ; jt += 2 * WPW) {
        loadB(jt + WPW, bB);             // prefetch tile 2 while computing tile 1
        compute(bA, jt * 16);
        const int jn = jt + 2 * WPW;
        loadB(jn < NTJ ? jn : wave, bA); // prefetch next pair (clamped: harmless reload)
        compute(bB, (jt + WPW) * 16);
    }

    // dump per-lane candidates
    const int rowloc = (halfsel ? 8 : 0);
#pragma unroll
    for (int r = 0; r < 8; ++r) {
        const int slot = wave * 80 + l15 * 5;
#pragma unroll
        for (int e = 0; e < 5; ++e) {
            sv[rowloc + r][slot + e] = tv[r][e];
            si[rowloc + r][slot + e] = ti[r][e];
        }
    }
    __syncthreads();

    // exact merge: one thread per row selects top-5 of 320 candidates
    if (threadIdx.x < 16) {
        const int row = threadIdx.x;
        float fv[5]; int fi[5];
#pragma unroll
        for (int e = 0; e < 5; ++e) { fv[e] = -1e30f; fi[e] = 0; }
        for (int q = 0; q < WPW * 16 * 5; ++q) insert5(fv, fi, sv[row][q], si[row][q]);
        float rs = fv[0] + fv[1] + fv[2] + fv[3] + fv[4];
        if (rs == 0.0f) rs = 1.0f;
        const float inv = 1.0f / rs;
        const size_t base = (size_t)(i0 + row) * 5;
#pragma unroll
        for (int e = 0; e < 5; ++e) {
            topk_idx[base + e] = fi[e];
            topk_w[base + e]   = fv[e] * inv;
        }
    }
}

// ============================================================
// 3) Generic WMMA GEMM:  C[M,Nn] = A[M,K] @ W[Nn,K]^T + bias (+relu)
//    f32 inputs converted to bf16 in registers; wave per 16x16 tile
//    K-loop software-pipelined (next chunk loads before current WMMA)
// ============================================================
__global__ __launch_bounds__(32) void gemm_kernel(const float* __restrict__ A, int lda,
                                                  const float* __restrict__ W, int ldw,
                                                  const float* __restrict__ bias,
                                                  float* __restrict__ C, int ldc,
                                                  int Kdim, int act) {
    const int lane = threadIdx.x & 31;
    const int m0   = blockIdx.x * 16;
    const int n0   = blockIdx.y * 16;
    const int l15  = lane & 15;
    const int half = lane >> 4;

    const float* arow = A + (size_t)(m0 + l15) * lda + (half ? 16 : 0);
    const float* wrow = W + (size_t)(n0 + l15) * ldw + (half ? 16 : 0);

    auto loadA = [&](int kb) -> bf16x16 {
        f32x8 a0 = *(const f32x8*)(arow + kb);
        f32x8 a1 = *(const f32x8*)(arow + kb + 8);
        return __builtin_shufflevector(__builtin_convertvector(a0, bf16x8),
                                       __builtin_convertvector(a1, bf16x8),
                                       0,1,2,3,4,5,6,7,8,9,10,11,12,13,14,15);
    };
    auto loadW = [&](int kb) -> bf16x16 {
        return __builtin_convertvector(*(const f32x16*)(wrow + kb), bf16x16);
    };

    f32x8 c = {0.f,0.f,0.f,0.f,0.f,0.f,0.f,0.f};
    bf16x16 aC = loadA(0), wC = loadW(0);
    for (int kb = 0; kb < Kdim; kb += 32) {
        const int kn = (kb + 32 < Kdim) ? kb + 32 : 0;   // clamped prefetch
        bf16x16 aN = loadA(kn);
        bf16x16 wN = loadW(kn);
        c = __builtin_amdgcn_wmma_f32_16x16x32_bf16(false, aC, false, wC,
                                                    (short)0, c, false, false);
        aC = aN; wC = wN;
    }

    const int n  = n0 + l15;
    const int mb = m0 + (half ? 8 : 0);
    const float bv = bias ? bias[n] : 0.0f;
#pragma unroll
    for (int r = 0; r < 8; ++r) {
        float v = c[r] + bv;
        if (act) v = fmaxf(v, 0.0f);
        C[(size_t)(mb + r) * ldc + n] = v;
    }
}

// ============================================================
// 4) LSTM gate nonlinearity:  h = sig(o)*tanh(sig(i)*tanh(g))
// ============================================================
__global__ __launch_bounds__(256) void lstm_gate_kernel(const float* __restrict__ g,
                                                        const float* __restrict__ bi,
                                                        const float* __restrict__ bh,
                                                        float* __restrict__ out,
                                                        int ostride, int ooff) {
    const int t = blockIdx.x * blockDim.x + threadIdx.x;
    if (t >= NROW * HDIM) return;
    const int n = t >> 6, c = t & 63;
    const float* gr = g + (size_t)n * 256;
    const float iv = gr[c]       + bi[c]       + bh[c];
    const float gv = gr[128 + c] + bi[128 + c] + bh[128 + c];
    const float ov = gr[192 + c] + bi[192 + c] + bh[192 + c];
    const float si = 1.0f / (1.0f + __expf(-iv));
    const float so = 1.0f / (1.0f + __expf(-ov));
    const float cc = si * tanhf(gv);
    out[(size_t)n * ostride + ooff + c] = so * tanhf(cc);
}

// ============================================================
// 5) LayerNorm over width 64, optional residual, strided output
// ============================================================
__global__ __launch_bounds__(256) void ln64_kernel(const float* __restrict__ in,
                                                   const float* __restrict__ res,
                                                   const float* __restrict__ g,
                                                   const float* __restrict__ b,
                                                   float* __restrict__ out,
                                                   int ostride, int ooff) {
    const int t = blockIdx.x * blockDim.x + threadIdx.x;
    if (t >= NROW) return;
    const float* ir = in + (size_t)t * HDIM;
    const float* rr = res ? res + (size_t)t * HDIM : nullptr;
    float s = 0.f, s2 = 0.f;
    for (int k = 0; k < HDIM; ++k) {
        const float x = ir[k] + (rr ? rr[k] : 0.0f);
        s += x; s2 += x * x;
    }
    const float m   = s * (1.0f / HDIM);
    const float var = s2 * (1.0f / HDIM) - m * m;
    const float inv = rsqrtf(var + 1e-5f);
    float* orow = out + (size_t)t * ostride + ooff;
    for (int k = 0; k < HDIM; ++k) {
        const float x = ir[k] + (rr ? rr[k] : 0.0f);
        orow[k] = (x - m) * inv * g[k] + b[k];
    }
}

// ============================================================
// 6) GNN aggregate: fused[:,192:256] = relu(sum_k w*feat[idx])
// ============================================================
__global__ __launch_bounds__(256) void gnn_agg_kernel(const float* __restrict__ feat,
                                                      const int* __restrict__ tidx,
                                                      const float* __restrict__ tw,
                                                      float* __restrict__ fused) {
    const int t = blockIdx.x * blockDim.x + threadIdx.x;
    if (t >= NROW * HDIM) return;
    const int n = t >> 6, c = t & 63;
    float acc = 0.f;
#pragma unroll
    for (int e = 0; e < 5; ++e)
        acc += tw[n * 5 + e] * feat[(size_t)tidx[n * 5 + e] * HDIM + c];
    fused[(size_t)n * 288 + 192 + c] = fmaxf(acc, 0.0f);
}

// ============================================================
// 7) Embedding gather + K-padding into fused[:,256:288]
// ============================================================
__global__ __launch_bounds__(256) void embed_kernel(const int* __restrict__ cid,
                                                    const int* __restrict__ iid,
                                                    const float* __restrict__ comp,
                                                    const float* __restrict__ ind,
                                                    float* __restrict__ fused) {
    const int t = blockIdx.x * blockDim.x + threadIdx.x;
    if (t >= NROW * 32) return;
    const int n = t >> 5, j = t & 31;
    float v;
    if (j < 16)      v = comp[(size_t)cid[n] * 16 + j];
    else if (j < 24) v = ind[(size_t)iid[n] * 8 + (j - 16)];
    else             v = 0.0f;   // zero-pad 280..287 so K=288 GEMM is exact
    fused[(size_t)n * 288 + 256 + j] = v;
}

// ============================================================
// 8) Pad fus1_W [128,280] -> [128,288] with zeros
// ============================================================
__global__ __launch_bounds__(256) void padw_kernel(const float* __restrict__ W,
                                                   float* __restrict__ Wp) {
    const int t = blockIdx.x * blockDim.x + threadIdx.x;
    if (t >= 128 * 288) return;
    const int r = t / 288, c = t % 288;
    Wp[t] = (c < 280) ? W[r * 280 + c] : 0.0f;
}

// ============================================================
// 9) Heads: distress [N], regime [N,4]
// ============================================================
__global__ __launch_bounds__(256) void heads_kernel(const float* __restrict__ f2,
                                                    const float* __restrict__ dW,
                                                    const float* __restrict__ db,
                                                    const float* __restrict__ rW,
                                                    const float* __restrict__ rb,
                                                    float* __restrict__ out) {
    const int t = blockIdx.x * blockDim.x + threadIdx.x;
    if (t >= NROW) return;
    const float* xr = f2 + (size_t)t * HDIM;
    float dd = db[0];
    float r0 = rb[0], r1 = rb[1], r2 = rb[2], r3 = rb[3];
    for (int k = 0; k < HDIM; ++k) {
        const float x = xr[k];
        dd += x * dW[k];
        r0 += x * rW[k]; r1 += x * rW[64 + k];
        r2 += x * rW[128 + k]; r3 += x * rW[192 + k];
    }
    out[t] = dd;
    float* rg = out + NROW + (size_t)t * 4;
    rg[0] = r0; rg[1] = r1; rg[2] = r2; rg[3] = r3;
}

// ============================================================
// host launcher
// ============================================================
extern "C" void kernel_launch(void* const* d_in, const int* in_sizes, int n_in,
                              void* d_out, int out_size, void* d_ws, size_t ws_size,
                              hipStream_t stream) {
    (void)in_sizes; (void)n_in; (void)out_size; (void)ws_size;

    // ---- inputs (setup_inputs dict order, depth-first) ----
    const float* X    = (const float*)d_in[0];
    const int*   cid  = (const int*)d_in[1];
    const int*   iid  = (const int*)d_in[2];
    // lstm: [3..8] layer0 fwd(W,bi,bh) bwd(W,bi,bh); [9..14] layer1
    // trans: [15..26] layer0, [27..38] layer1
    const float* projW = (const float*)d_in[39];
    const float* projb = (const float*)d_in[40];
    const float* lng   = (const float*)d_in[41];
    const float* lnb   = (const float*)d_in[42];
    const float* gnnW  = (const float*)d_in[43];
    const float* gnnb  = (const float*)d_in[44];
    const float* cemb  = (const float*)d_in[45];
    const float* iemb  = (const float*)d_in[46];
    const float* f1W   = (const float*)d_in[47];
    const float* f1b   = (const float*)d_in[48];
    const float* f2W   = (const float*)d_in[49];
    const float* f2b   = (const float*)d_in[50];
    const float* disW  = (const float*)d_in[51];
    const float* disb  = (const float*)d_in[52];
    const float* regW  = (const float*)d_in[53];
    const float* regb  = (const float*)d_in[54];
    float* out = (float*)d_out;

    // ---- workspace carve ----
    char* ws = (char*)d_ws;
    size_t off = 0;
    auto carve = [&](size_t bytes) -> void* {
        void* p = ws + off;
        off = (off + bytes + 255) & ~(size_t)255;
        return p;
    };
    __bf16* Xnb  = (__bf16*)carve((size_t)NROW * DIMD * 2);
    float*  g0   = (float*) carve((size_t)NROW * 256 * 4);
    float*  g1   = (float*) carve((size_t)NROW * 256 * 4);
    float*  hbuf = (float*) carve((size_t)NROW * 128 * 4);
    float*  t0   = (float*) carve((size_t)NROW * 64 * 4);
    float*  t1   = (float*) carve((size_t)NROW * 64 * 4);
    float*  t2   = (float*) carve((size_t)NROW * 64 * 4);
    float*  ffb  = (float*) carve((size_t)NROW * 128 * 4);
    float*  feat = (float*) carve((size_t)NROW * 64 * 4);
    float*  fusd = (float*) carve((size_t)NROW * 288 * 4);
    int*    tidx = (int*)   carve((size_t)NROW * 5 * 4);
    float*  tw   = (float*) carve((size_t)NROW * 5 * 4);
    float*  wpad = (float*) carve((size_t)128 * 288 * 4);

    auto gemm = [&](const float* A, int lda, const float* W, int ldw,
                    const float* bias, float* C, int ldc, int Kd, int Nn, int act) {
        dim3 grid(NROW / 16, Nn / 16);
        gemm_kernel<<<grid, 32, 0, stream>>>(A, lda, W, ldw, bias, C, ldc, Kd, act);
    };

    // ===== GNN branch: kNN graph (hotspot) =====
    rownorm_kernel<<<NROW / 4, 128, 0, stream>>>(X, Xnb);
    sim_topk_kernel<<<NROW / 16, 128, 0, stream>>>(Xnb, tidx, tw);
    gemm(X, DIMD, gnnW, DIMD, gnnb, feat, 64, DIMD, 64, 0);     // feat = X@gnnW.T+b
    gnn_agg_kernel<<<(NROW * 64) / 256, 256, 0, stream>>>(feat, tidx, tw, fusd);

    // ===== LSTM branch =====
    // layer 0 (K=512) -> hbuf[N,128]
    gemm(X, DIMD, (const float*)d_in[3], DIMD, nullptr, g0, 256, DIMD, 256, 0);
    gemm(X, DIMD, (const float*)d_in[6], DIMD, nullptr, g1, 256, DIMD, 256, 0);
    lstm_gate_kernel<<<(NROW * 64) / 256, 256, 0, stream>>>(g0,
        (const float*)d_in[4], (const float*)d_in[5], hbuf, 128, 0);
    lstm_gate_kernel<<<(NROW * 64) / 256, 256, 0, stream>>>(g1,
        (const float*)d_in[7], (const float*)d_in[8], hbuf, 128, 64);
    // layer 1 (K=128) -> fused[:,0:128]
    gemm(hbuf, 128, (const float*)d_in[9],  128, nullptr, g0, 256, 128, 256, 0);
    gemm(hbuf, 128, (const float*)d_in[12], 128, nullptr, g1, 256, 128, 256, 0);
    lstm_gate_kernel<<<(NROW * 64) / 256, 256, 0, stream>>>(g0,
        (const float*)d_in[10], (const float*)d_in[11], fusd, 288, 0);
    lstm_gate_kernel<<<(NROW * 64) / 256, 256, 0, stream>>>(g1,
        (const float*)d_in[13], (const float*)d_in[14], fusd, 288, 64);

    // ===== Transformer branch =====
    gemm(X, DIMD, projW, DIMD, projb, t0, 64, DIMD, 64, 0);     // t0 = proj(X)
    for (int L = 0; L < 2; ++L) {
        const int base = 15 + 12 * L;
        const float* Wqkv = (const float*)d_in[base + 0];
        const float* bqkv = (const float*)d_in[base + 1];
        const float* Wo   = (const float*)d_in[base + 2];
        const float* bo   = (const float*)d_in[base + 3];
        const float* W1   = (const float*)d_in[base + 4];
        const float* b1   = (const float*)d_in[base + 5];
        const float* W2   = (const float*)d_in[base + 6];
        const float* b2   = (const float*)d_in[base + 7];
        const float* l1g  = (const float*)d_in[base + 8];
        const float* l1b  = (const float*)d_in[base + 9];
        const float* l2g  = (const float*)d_in[base + 10];
        const float* l2b  = (const float*)d_in[base + 11];
        // v = t0 @ Wv.T + bv  (Wv = Wqkv rows [128:192])
        gemm(t0, 64, Wqkv + 128 * 64, 64, bqkv + 128, t1, 64, 64, 64, 0);
        // attn = v @ Wo.T + bo
        gemm(t1, 64, Wo, 64, bo, t2, 64, 64, 64, 0);
        // t0 = LN(t0 + attn)
        ln64_kernel<<<NROW / 256, 256, 0, stream>>>(t0, t2, l1g, l1b, t0, 64, 0);
        // ff = relu(t0@W1.T+b1) @ W2.T + b2
        gemm(t0, 64, W1, 64, b1, ffb, 128, 64, 128, 1);
        gemm(ffb, 128, W2, 128, b2, t2, 64, 128, 64, 0);
        // t0 = LN(t0 + ff)
        ln64_kernel<<<NROW / 256, 256, 0, stream>>>(t0, t2, l2g, l2b, t0, 64, 0);
    }
    // trans_out = LN(t0) -> fused[:,128:192]
    ln64_kernel<<<NROW / 256, 256, 0, stream>>>(t0, nullptr, lng, lnb, fusd, 288, 128);

    // ===== embeddings + pad -> fused[:,256:288] =====
    embed_kernel<<<(NROW * 32) / 256, 256, 0, stream>>>(cid, iid, cemb, iemb, fusd);

    // ===== fusion MLP =====
    padw_kernel<<<(128 * 288 + 255) / 256, 256, 0, stream>>>(f1W, wpad);
    gemm(fusd, 288, wpad, 288, f1b, ffb, 128, 288, 128, 1);     // relu
    gemm(ffb, 128, f2W, 128, f2b, t0, 64, 128, 64, 1);          // relu

    // ===== heads =====
    heads_kernel<<<NROW / 256, 256, 0, stream>>>(t0, disW, disb, regW, regb, out);
}
